// MACEModel_69887707841292
// MI455X (gfx1250) — matrix operand order repaired and checked by reading, not compiled, not run on hardware
//
#include <hip/hip_runtime.h>

// ---------------------------------------------------------------------------
// MACE forward for MI455X (gfx1250).
// - Heavy GEMMs -> v_wmma_f32_16x16x32_f16 (f16 in, f32 accum).
// - All WMMA operands pre-packed into ISA 7.12.2 per-lane fragment order so
//   every fragment load is one contiguous 32B (2 x b128) access.
// - MLP-2 GEMM fused with tp_conv: the 738MB per-edge weight tensor never
//   touches HBM; 16-edge x 1024-col weight tiles live in LDS (~84KB/WG).
// - Hidden tile staged into LDS via the Tensor Data Mover (tensor_load_to_lds
//   + s_wait_tensorcnt) when the builtin is available.
// ---------------------------------------------------------------------------

#define NNODE 8192
#define EIN   8192
#define ETOT  16384
#define DIN   64
#define CCH   32
#define MLPD  256
#define NCMP  9      // 1 + 3 + 5 irreps components

typedef __attribute__((ext_vector_type(16))) _Float16     v16h;
typedef __attribute__((ext_vector_type(8)))  float        v8f;
typedef __attribute__((ext_vector_type(4)))  unsigned int v4u;
typedef __attribute__((ext_vector_type(8)))  int          v8i;
typedef __attribute__((ext_vector_type(4)))  int          v4i;

#if defined(__has_builtin)
#if __has_builtin(__builtin_amdgcn_tensor_load_to_lds) && \
    __has_builtin(__builtin_amdgcn_s_wait_tensorcnt)
#define HAVE_TDM 1
#endif
#endif
#ifndef HAVE_TDM
#define HAVE_TDM 0
#endif

// ---------------- CG / path constant tables --------------------------------
// Global PATHS order (l1-major, matching the reference iteration order):
//  0:(0,0,0) 1:(0,1,1) 2:(0,2,2) 3:(1,0,1) 4:(1,1,0) 5:(1,1,2)
//  6:(1,2,1) 7:(2,0,2) 8:(2,1,1) 9:(2,2,0) 10:(2,2,2)
__constant__ int c_pl1[11] = {0,0,0,1,1,1,1,2,2,2,2};
__constant__ int c_pl2[11] = {0,1,2,0,1,1,2,0,1,2,2};
__constant__ int c_pl3[11] = {0,1,2,1,0,2,1,2,1,0,2};
__constant__ int c_off[3]  = {0,1,4};
__constant__ int c_ld[3]   = {1,3,5};
__constant__ int c_cgs[12] = {0,1,4,9,12,15,26,37,42,53,58,83};

__constant__ unsigned char c_cgi[83] = {
  0,
  0,0,0,
  0,0,0,0,0,
  0,1,2,
  0,1,2,
  2,0,0,1,0,2,1,2,1,2,0,
  2,0,0,1,0,2,1,2,1,2,0,
  0,1,2,3,4,
  0,0,1,1,2,2,2,3,3,4,4,
  0,1,2,3,4,
  0,0,2,1,1,2,3,3,2,4,4,2,1,1,4,3,3,4,0,0,1,1,3,3,2
};
__constant__ unsigned char c_cgj[83] = {
  0,
  0,1,2,
  0,1,2,3,4,
  0,0,0,
  0,1,2,
  0,2,1,0,0,2,1,1,2,2,0,
  0,0,1,1,2,2,2,3,3,4,4,
  0,0,0,0,0,
  2,0,0,1,0,2,1,2,1,2,0,
  0,1,2,3,4,
  0,2,0,1,2,1,3,2,3,4,2,4,1,4,1,3,4,3,1,3,0,3,0,1,2
};
__constant__ unsigned char c_cgk[83] = {
  0,
  0,1,2,
  0,1,2,3,4,
  0,1,2,
  0,0,0,
  0,0,1,1,2,2,2,3,3,4,4,
  0,2,1,0,0,2,1,1,2,2,0,
  0,1,2,3,4,
  0,2,1,0,0,2,1,1,2,2,0,
  0,0,0,0,0,
  2,0,0,2,1,1,2,3,3,2,4,4,4,1,1,4,3,3,3,1,3,0,1,0,2
};
__constant__ float c_cgv[83] = {
  1.0f,
  0.57735027f,0.57735027f,0.57735027f,
  0.44721360f,0.44721360f,0.44721360f,0.44721360f,0.44721360f,
  0.57735027f,0.57735027f,0.57735027f,
  0.57735027f,0.57735027f,0.57735027f,
  0.31622777f,0.31622777f,0.31622777f,0.31622777f,-0.18257419f,-0.18257419f,
  0.36514837f,0.31622777f,0.31622777f,0.31622777f,-0.31622777f,
  0.31622777f,0.31622777f,0.31622777f,0.31622777f,-0.18257419f,-0.18257419f,
  0.36514837f,0.31622777f,0.31622777f,0.31622777f,-0.31622777f,
  0.44721360f,0.44721360f,0.44721360f,0.44721360f,0.44721360f,
  0.31622777f,0.31622777f,0.31622777f,0.31622777f,-0.18257419f,-0.18257419f,
  0.36514837f,0.31622777f,0.31622777f,0.31622777f,-0.31622777f,
  0.44721360f,0.44721360f,0.44721360f,0.44721360f,0.44721360f,
  -0.23904572f,-0.23904572f,-0.23904572f, 0.11952286f, 0.11952286f, 0.11952286f,
   0.11952286f, 0.11952286f, 0.11952286f,-0.23904572f,-0.23904572f,-0.23904572f,
  -0.20701967f,-0.20701967f,-0.20701967f, 0.20701967f, 0.20701967f, 0.20701967f,
   0.20701967f, 0.20701967f, 0.20701967f, 0.20701967f, 0.20701967f, 0.20701967f,
   0.23904572f
};

// ---------------- WMMA helper ----------------------------------------------
__device__ inline v8f wmma_f16(v16h a, v16h b, v8f c) {
  return __builtin_amdgcn_wmma_f32_16x16x32_f16(false, a, false, b, (short)0, c,
                                                false, false);
}
__device__ inline v16h ld_frag(const _Float16* p) { return *(const v16h*)p; }

// Fragment packing index helpers (ISA 7.12.2, f16 16x16x32):
// A (16x32, M x K): lane = m + (k&8 ? 16:0), elem = (k&7) + ((k&16)>>1)
// B (32x16, K x N): lane = n + (k&8 ? 16:0), elem = (k&7) + ((k&16)>>1)

// ---------------- packing kernels ------------------------------------------
// f32 row-major [M][Kin] -> f16 A-fragment order (K zero-padded to K)
__global__ void k_pack_a(const float* __restrict__ in, _Float16* __restrict__ out,
                         int M, int Kin, int K) {
  int i = blockIdx.x * 256 + threadIdx.x;
  if (i >= M * K) return;
  int row = i / K, col = i % K;
  float v = (col < Kin) ? in[(size_t)row * Kin + col] : 0.0f;
  int mt = row >> 4, m = row & 15, ks = col >> 5, kk = col & 31;
  int lane = m + ((kk & 8) ? 16 : 0);
  int elem = (kk & 7) + ((kk & 16) >> 1);
  out[(((size_t)mt * (K >> 5) + ks) * 32 + lane) * 16 + elem] = (_Float16)v;
}

// f32 row-major [Kin][N] -> f16 B-fragment order (K zero-padded to K)
__global__ void k_pack_b(const float* __restrict__ in, _Float16* __restrict__ out,
                         int Kin, int K, int N) {
  int i = blockIdx.x * 256 + threadIdx.x;
  if (i >= K * N) return;
  int k = i / N, n = i % N;
  float v = (k < Kin) ? in[(size_t)k * N + n] : 0.0f;
  int ks = k >> 5, kk = k & 31, nt = n >> 4, nc = n & 15;
  int lane = nc + ((kk & 8) ? 16 : 0);
  int elem = (kk & 7) + ((kk & 16) >> 1);
  out[(((size_t)nt * (K >> 5) + ks) * 32 + lane) * 16 + elem] = (_Float16)v;
}

__global__ void k_zero_f32(float* p, int n) {
  int i = blockIdx.x * 256 + threadIdx.x;
  if (i < n) p[i] = 0.0f;
}
__global__ void k_zero_u32(unsigned* p, int n) {
  int i = blockIdx.x * 256 + threadIdx.x;
  if (i < n) p[i] = 0u;
}

// ---------------- node embedding: h0 = x @ Wemb + b ------------------------
__global__ __launch_bounds__(128) void k_node_embed(
    const _Float16* __restrict__ xp, const _Float16* __restrict__ wp,
    const float* __restrict__ bias, float* __restrict__ h) {
  const int wg   = blockIdx.x * 4 + (threadIdx.x >> 5);
  const int lane = threadIdx.x & 31;
  const int mt = wg >> 1, nt = wg & 1;          // 512 x 2 tiles, K=64
  const _Float16* ap = xp + (((size_t)mt * 2) * 32 + lane) * 16;
  const _Float16* bp = wp + (((size_t)nt * 2) * 32 + lane) * 16;
  v8f acc = {};
#pragma unroll
  for (int ks = 0; ks < 2; ++ks)
    acc = wmma_f16(ld_frag(ap + ks * 512), ld_frag(bp + ks * 512), acc);
  const int rbase = mt * 16 + ((lane & 16) ? 8 : 0);
  const int col   = nt * 16 + (lane & 15);
#pragma unroll
  for (int v = 0; v < 8; ++v) {
    float* hp = h + ((size_t)(rbase + v) * CCH + col) * NCMP;
    hp[0] = acc[v] + bias[col];
#pragma unroll
    for (int q = 1; q < NCMP; ++q) hp[q] = 0.0f;
  }
}

// ---------------- edge geometry: SH + Bessel -------------------------------
__global__ void k_edge_geom(const float* __restrict__ pos, const int* __restrict__ ei,
                            int* __restrict__ srcv, int* __restrict__ dstv,
                            float* __restrict__ shb, float* __restrict__ radb) {
  int e = blockIdx.x * 256 + threadIdx.x;
  if (e >= ETOT) return;
  int s, d;
  if (e < EIN) { s = ei[e]; d = ei[EIN + e]; }
  else         { s = ei[e]; d = ei[e - EIN]; }   // mirrored edges
  srcv[e] = s; dstv[e] = d;
  float vx = pos[s * 3 + 0] - pos[d * 3 + 0];
  float vy = pos[s * 3 + 1] - pos[d * 3 + 1];
  float vz = pos[s * 3 + 2] - pos[d * 3 + 2];
  float r  = sqrtf(vx * vx + vy * vy + vz * vz);
  float iv = 1.0f / fmaxf(r, 1e-9f);
  float x = vx * iv, y = vy * iv, z = vz * iv;
  float* sh = shb + (size_t)e * NCMP;
  sh[0] = 1.0f;
  sh[1] = 1.7320508f * y; sh[2] = 1.7320508f * z; sh[3] = 1.7320508f * x;
  sh[4] = 3.8729833f * x * y;
  sh[5] = 3.8729833f * y * z;
  sh[6] = 1.1180340f * (3.0f * z * z - 1.0f);
  sh[7] = 3.8729833f * x * z;
  sh[8] = 1.9364917f * (x * x - y * y);
  float u  = r * 0.1f;
  float u5 = u * u * u * u * u;
  float env = (u < 1.0f) ? (1.0f - 21.0f * u5 + 35.0f * u5 * u - 15.0f * u5 * u * u)
                         : 0.0f;
  float rinv = 1.0f / r;
#pragma unroll
  for (int n = 1; n <= 8; ++n)
    radb[(size_t)e * 8 + n - 1] =
        0.4472136f * sinf((float)n * 3.14159265f * r * 0.1f) * rinv * env;
}

__global__ void k_count(const int* __restrict__ dstv, unsigned* __restrict__ cnt) {
  int e = blockIdx.x * 256 + threadIdx.x;
  if (e < ETOT) atomicAdd(&cnt[dstv[e]], 1u);
}

// ---------------- edge MLP input, written in A-fragment order --------------
__global__ void k_edge_feat(const float* __restrict__ radb,
                            const float* __restrict__ hin,
                            const int* __restrict__ srcv, const int* __restrict__ dstv,
                            _Float16* __restrict__ efp) {
  int i = blockIdx.x * 256 + threadIdx.x;
  if (i >= ETOT * 96) return;
  int e = i / 96, col = i % 96;
  float v;
  if (col < 8)       v = radb[(size_t)e * 8 + col];
  else if (col < 40) v = hin[((size_t)srcv[e] * CCH + (col - 8)) * NCMP];
  else if (col < 72) v = hin[((size_t)dstv[e] * CCH + (col - 40)) * NCMP];
  else               v = 0.0f;
  int mt = e >> 4, m = e & 15, ks = col >> 5, kk = col & 31;
  int lane = m + ((kk & 8) ? 16 : 0);
  int elem = (kk & 7) + ((kk & 16) >> 1);
  efp[(((size_t)mt * 3 + ks) * 32 + lane) * 16 + elem] = (_Float16)v;
}

// ---------------- edge MLP layer 1 (WMMA, K=96, relu) ----------------------
// Output written in A-fragment order (K=256) for the fused MLP-2 kernel.
__global__ __launch_bounds__(128) void k_mlp1(
    const _Float16* __restrict__ efp, const _Float16* __restrict__ w1p,
    const float* __restrict__ b1, _Float16* __restrict__ hidp) {
  const int wg   = blockIdx.x * 4 + (threadIdx.x >> 5);
  const int lane = threadIdx.x & 31;
  const int mt = wg >> 4, nt = wg & 15;          // 1024 x 16 tiles
  const _Float16* ap = efp + (((size_t)mt * 3) * 32 + lane) * 16;
  const _Float16* bp = w1p + (((size_t)nt * 3) * 32 + lane) * 16;
  v8f acc = {};
#pragma unroll
  for (int ks = 0; ks < 3; ++ks)
    acc = wmma_f16(ld_frag(ap + ks * 512), ld_frag(bp + ks * 512), acc);
  const int rbase = (lane & 16) ? 8 : 0;
  const int col   = nt * 16 + (lane & 15);
  const int ks2 = col >> 5, kk = col & 31;
  const int lhalf = (kk & 8) ? 16 : 0;
  const int elem  = (kk & 7) + ((kk & 16) >> 1);
#pragma unroll
  for (int v = 0; v < 8; ++v) {
    float o = acc[v] + b1[col];
    int m = rbase + v;
    hidp[(((size_t)mt * 8 + ks2) * 32 + m + lhalf) * 16 + elem] =
        (_Float16)(o > 0.0f ? o : 0.0f);
  }
}

// ---------------- fused MLP-2 GEMM + tensor-product conv -------------------
// Block = 16 edges, 4 waves. The 16x256 hidden tile (contiguous 8KB in
// fragment-packed layout) is staged into LDS via the Tensor Data Mover.
__global__ __launch_bounds__(128) void k_mlp2_tpconv(
    const _Float16* __restrict__ hidp, const _Float16* __restrict__ w2p,
    const float* __restrict__ b2, const float* __restrict__ hin,
    const float* __restrict__ shb, const int* __restrict__ srcv,
    const int* __restrict__ dstv, float* __restrict__ msgs,
    int numel, int npaths) {
  __shared__ __attribute__((aligned(32))) _Float16 s_hid[16 * MLPD]; // 8 KB
  __shared__ float s_w[16][1024];                                    // 64 KB
  __shared__ float s_t[16][CCH][5];                                  // 10 KB
  __shared__ int   s_src[16], s_dst[16];

  const int tid  = threadIdx.x;
  const int lane = tid & 31, wid = tid >> 5;
  const int e0   = blockIdx.x * 16;

#if HAVE_TDM
  if (wid == 0) {
    // Tensor DMA descriptor: 1D copy of 4096 f16 elements (8KB) -> LDS.
    unsigned lds_a = (unsigned)(uintptr_t)&s_hid[0];
    unsigned long long ga =
        (unsigned long long)(uintptr_t)(hidp + (size_t)e0 * MLPD);
    v4u g0;
    g0[0] = 1u;                                     // count=1, normal mode
    g0[1] = lds_a;                                  // lds_addr
    g0[2] = (unsigned)(ga & 0xffffffffu);           // global_addr[31:0]
    g0[3] = (unsigned)((ga >> 32) & 0x01ffffffu) | 0x80000000u; // type=2
    v8i g1;
    g1[0] = 0x00010000;             // data_size = 2 bytes
    g1[1] = (int)(4096u << 16);     // tensor_dim0 = 4096
    g1[2] = 0x00010000;             // tensor_dim1 = 1
    g1[3] = (int)(4096u << 16);     // tile_dim0   = 4096
    g1[4] = 1;                      // tile_dim1   = 1
    g1[5] = 4096;                   // tensor_dim0_stride
    g1[6] = 0; g1[7] = 0;
    v4i gz = {0, 0, 0, 0};
#if __has_include(<hip/amd_detail/amd_gfx1250_TDM.h>)
    v8i z8 = {0, 0, 0, 0, 0, 0, 0, 0};
    __builtin_amdgcn_tensor_load_to_lds(g0, g1, gz, gz, z8, 0);
#else
    __builtin_amdgcn_tensor_load_to_lds(g0, g1, gz, gz, 0);
#endif
    __builtin_amdgcn_s_wait_tensorcnt(0);
  }
#else
  for (int i = tid; i < 16 * MLPD; i += 128)
    s_hid[i] = hidp[(size_t)e0 * MLPD + i];
#endif
  if (tid < 16) { s_src[tid] = srcv[e0 + tid]; s_dst[tid] = dstv[e0 + tid]; }
  __syncthreads();

  for (int p = 0; p < npaths; ++p) {
    const int l1 = c_pl1[p], l2 = c_pl2[p], l3 = c_pl3[p];
    const int io = c_off[l1], jo = c_off[l2], ko = c_off[l3];
    const int ld3 = c_ld[l3];
    const int off = p * 1024;

    // phase 1: w_tile = hidden_tile @ W2[:, off:off+1024] + b2  (WMMA)
    for (int nt = wid; nt < 64; nt += 4) {
      v8f acc = {};
      const _Float16* bp = w2p + (((size_t)(p * 64 + nt) * 8) * 32 + lane) * 16;
#pragma unroll
      for (int ks = 0; ks < 8; ++ks) {
        if (ks < 7) __builtin_prefetch(bp + (ks + 1) * 512, 0, 1);
        v16h a = ld_frag(s_hid + ((ks * 32 + lane) << 4));
        v16h b = ld_frag(bp + ks * 512);
        acc = wmma_f16(a, b, acc);
      }
      const int rbase = (lane & 16) ? 8 : 0;
      const int col   = nt * 16 + (lane & 15);
#pragma unroll
      for (int v = 0; v < 8; ++v)
        s_w[rbase + v][col] = acc[v] + b2[off + col];
    }
    __syncthreads();

    // phase 2: t[e,c,k] = sum_{ij} h[l1][src][c][i] * sh[l2][e][j] * CG[i,j,k]
    for (int i = tid; i < 512; i += 128) {
      const int el = i >> 5, c = i & 31;
      const float* hv = hin + ((size_t)s_src[el] * CCH + c) * NCMP + io;
      const float* sv = shb + (size_t)(e0 + el) * NCMP + jo;
      float tk[5] = {0, 0, 0, 0, 0};
      for (int n = c_cgs[p]; n < c_cgs[p + 1]; ++n)
        tk[c_cgk[n]] += hv[c_cgi[n]] * sv[c_cgj[n]] * c_cgv[n];
      for (int k = 0; k < ld3; ++k) s_t[el][c][k] = tk[k];
    }
    __syncthreads();

    // phase 3: m[e,d,k] = sum_c t[e,c,k] * w[e,c,d] / sqrt(C), scatter to dst
    for (int i = tid; i < 512; i += 128) {
      const int el = i >> 5, d = i & 31;
      float acc[5] = {0, 0, 0, 0, 0};
      for (int c2 = 0; c2 < 32; ++c2) {
        const float wv = s_w[el][c2 * 32 + d];
        for (int k = 0; k < ld3; ++k) acc[k] += s_t[el][c2][k] * wv;
      }
      float* mp = msgs + ((size_t)s_dst[el] * CCH + d) * NCMP + ko;
      for (int k = 0; k < ld3; ++k) atomicAdd(mp + k, acc[k] * 0.1767766953f);
    }
    __syncthreads();
  }
}

// ---------------- aggregation scaling --------------------------------------
__global__ void k_agg_finish(float* __restrict__ msgs,
                             const unsigned* __restrict__ cnt,
                             float i0, float i1, float i2) {
  int i = blockIdx.x * 256 + threadIdx.x;
  if (i >= NNODE * CCH * NCMP) return;
  int n = i / (CCH * NCMP);
  int q = i % NCMP;
  float inv = (q == 0) ? i0 : (q < 4 ? i1 : i2);
  float c = fmaxf((float)cnt[n], 1.0f);
  msgs[i] = msgs[i] / c * inv;
}

// ---------------- equivariant batch norm -----------------------------------
__global__ __launch_bounds__(256) void k_bn_stats(const float* __restrict__ h,
                                                  float* __restrict__ stats) {
  __shared__ float r0[256], r1[256], r2[256], r3[256];
  const int c = blockIdx.x, tid = threadIdx.x;
  float s0 = 0, sq = 0, s1 = 0, s2 = 0;
  for (int n = tid; n < NNODE; n += 256) {
    const float* v = h + ((size_t)n * CCH + c) * NCMP;
    float a = v[0];
    s0 += a; sq += a * a;
    s1 += (v[1] * v[1] + v[2] * v[2] + v[3] * v[3]) * (1.0f / 3.0f);
    s2 += (v[4] * v[4] + v[5] * v[5] + v[6] * v[6] + v[7] * v[7] + v[8] * v[8]) * 0.2f;
  }
  r0[tid] = s0; r1[tid] = sq; r2[tid] = s1; r3[tid] = s2;
  __syncthreads();
  for (int s = 128; s > 0; s >>= 1) {
    if (tid < s) {
      r0[tid] += r0[tid + s]; r1[tid] += r1[tid + s];
      r2[tid] += r2[tid + s]; r3[tid] += r3[tid + s];
    }
    __syncthreads();
  }
  if (tid == 0) {
    stats[c * 4 + 0] = r0[0]; stats[c * 4 + 1] = r1[0];
    stats[c * 4 + 2] = r2[0]; stats[c * 4 + 3] = r3[0];
  }
}

__global__ void k_bn_apply(float* __restrict__ h, const float* __restrict__ stats,
                           int nl) {
  int i = blockIdx.x * 256 + threadIdx.x;
  if (i >= NNODE * CCH) return;
  int c = i % CCH;
  float* v = h + (size_t)i * NCMP;
  const float invN = 1.0f / (float)NNODE;
  float mu  = stats[c * 4 + 0] * invN;
  float var = stats[c * 4 + 1] * invN - mu * mu;
  v[0] = (v[0] - mu) * rsqrtf(var + 1e-5f);
  if (nl > 1) {
    float f1 = rsqrtf(stats[c * 4 + 2] * invN + 1e-5f);
    v[1] *= f1; v[2] *= f1; v[3] *= f1;
    float f2 = rsqrtf(stats[c * 4 + 3] * invN + 1e-5f);
    v[4] *= f2; v[5] *= f2; v[6] *= f2; v[7] *= f2; v[8] *= f2;
  }
}

// ---------------- product block (symmetric contraction + linear) -----------
__global__ __launch_bounds__(32) void k_product(
    const float* __restrict__ hu, const float* __restrict__ sc,
    const float* __restrict__ w1, const float* __restrict__ cw2,
    const float* __restrict__ cw3, const float* __restrict__ lin,
    float* __restrict__ out, int nl, int use_sc) {
  __shared__ float cmb[CCH][NCMP];
  const int n = blockIdx.x, c = threadIdx.x;
  const float* Ap = hu + ((size_t)n * CCH + c) * NCMP;
  float A[NCMP], B2[NCMP], B3[NCMP];
#pragma unroll
  for (int q = 0; q < NCMP; ++q) { A[q] = Ap[q]; B2[q] = 0; B3[q] = 0; }
  const float is11 = 0.30151134f;  // 1/sqrt(11)
  for (int p = 0; p < 11; ++p) {
    const int io = c_off[c_pl1[p]], jo = c_off[c_pl2[p]], ko = c_off[c_pl3[p]];
    const float wpc = cw2[p * CCH + c] * is11;
    for (int t = c_cgs[p]; t < c_cgs[p + 1]; ++t)
      B2[ko + c_cgk[t]] += A[io + c_cgi[t]] * A[jo + c_cgj[t]] * c_cgv[t] * wpc;
  }
  for (int p = 0; p < 11; ++p) {
    const int io = c_off[c_pl1[p]], jo = c_off[c_pl2[p]], ko = c_off[c_pl3[p]];
    const float wpc = cw3[p * CCH + c] * is11;
    for (int t = c_cgs[p]; t < c_cgs[p + 1]; ++t)
      B3[ko + c_cgk[t]] += B2[io + c_cgi[t]] * A[jo + c_cgj[t]] * c_cgv[t] * wpc;
  }
#pragma unroll
  for (int q = 0; q < NCMP; ++q) cmb[c][q] = 0.0f;
  for (int t = 0; t < nl; ++t) {
    const int o = c_off[t], ld = c_ld[t];
    for (int i = 0; i < ld; ++i)
      cmb[c][o + i] = w1[t * CCH + c] * A[o + i] + B2[o + i] + B3[o + i];
  }
  __syncthreads();
  const int d = c;
  for (int t = 0; t < nl; ++t) {
    const int o = c_off[t], ld = c_ld[t];
    for (int i = 0; i < ld; ++i) {
      float s = 0.0f;
      for (int cc = 0; cc < CCH; ++cc)
        s += cmb[cc][o + i] * lin[(size_t)(t * CCH + cc) * CCH + d];
      s *= 0.1767766953f;  // 1/sqrt(C)
      if (use_sc && t == 0) s += sc[((size_t)n * CCH + d) * NCMP];
      out[((size_t)n * CCH + d) * NCMP + o + i] = s;
    }
  }
  if (nl == 1)
    for (int q = 1; q < NCMP; ++q) out[((size_t)n * CCH + d) * NCMP + q] = 0.0f;
}

// ---------------- readout ---------------------------------------------------
__global__ void k_final_node(const float* __restrict__ h, float* __restrict__ out) {
  int i = blockIdx.x * 256 + threadIdx.x;
  if (i < NNODE * CCH) out[i] = h[(size_t)i * NCMP];
}
__global__ void k_final_graph(const float* __restrict__ h,
                              const int* __restrict__ batch,
                              float* __restrict__ out) {
  int i = blockIdx.x * 256 + threadIdx.x;
  if (i >= NNODE * CCH) return;
  int n = i / CCH, c = i % CCH;
  atomicAdd(&out[NNODE * CCH + batch[n] * CCH + c], h[(size_t)i * NCMP]);
}

// ---------------------------------------------------------------------------
extern "C" void kernel_launch(void* const* d_in, const int* in_sizes, int n_in,
                              void* d_out, int out_size, void* d_ws, size_t ws_size,
                              hipStream_t stream) {
  // --- resolve inputs by (flat size, occurrence order): robust to pytree order
  const float *x = 0, *pos = 0, *embW = 0, *embB = 0;
  const int *ei = 0, *batch = 0;
  const float *W1[2] = {0, 0}, *W2[2] = {0, 0}, *B1[2] = {0, 0}, *B2p[2] = {0, 0};
  const float *LIN[2] = {0, 0}, *PW1[2] = {0, 0}, *CW2[2] = {0, 0}, *CW3[2] = {0, 0};
  int o32 = 0, o18432 = 0, o256 = 0, o3072 = 0, o352 = 0;
  for (int i = 0; i < n_in; ++i) {
    const void* p = d_in[i];
    switch (in_sizes[i]) {
      case 524288:  x = (const float*)p; break;
      case 24576:   pos = (const float*)p; break;
      case 16384:   ei = (const int*)p; break;
      case 8192:    batch = (const int*)p; break;
      case 2048:    embW = (const float*)p; break;
      case 32:      if (o32++ == 0) embB = (const float*)p; else PW1[1] = (const float*)p; break;
      case 18432:   if (o18432 < 2) W1[o18432++] = (const float*)p; break;
      case 786432:  W2[0] = (const float*)p; break;
      case 2883584: W2[1] = (const float*)p; break;
      case 256:     if (o256 < 2) B1[o256++] = (const float*)p; break;
      case 3072:    if (o3072++ == 0) B2p[0] = (const float*)p; else LIN[0] = (const float*)p; break;
      case 11264:   B2p[1] = (const float*)p; break;
      case 1024:    LIN[1] = (const float*)p; break;
      case 96:      PW1[0] = (const float*)p; break;
      case 352:
        if (o352 == 0) CW2[0] = (const float*)p;
        else if (o352 == 1) CW3[0] = (const float*)p;
        else if (o352 == 2) CW2[1] = (const float*)p;
        else CW3[1] = (const float*)p;
        ++o352; break;
      default: break;
    }
  }
  if (!x || !pos || !ei || !batch || !embW || !embB || !W1[1] || !W2[1]) return;

  // --- carve workspace (~48 MB)
  char* wsb = (char*)d_ws;
  size_t off = 0;
  auto alloc = [&](size_t bytes) -> void* {
    void* r = wsb + off;
    off += (bytes + 255) & ~(size_t)255;
    return r;
  };
  _Float16* xp   = (_Float16*)alloc((size_t)NNODE * DIN * 2);
  _Float16* ewp  = (_Float16*)alloc((size_t)DIN * CCH * 2);
  float* hA   = (float*)alloc((size_t)NNODE * CCH * NCMP * 4);
  float* hC   = (float*)alloc((size_t)NNODE * CCH * NCMP * 4);
  float* msgs = (float*)alloc((size_t)NNODE * CCH * NCMP * 4);
  float* shb  = (float*)alloc((size_t)ETOT * NCMP * 4);
  float* radb = (float*)alloc((size_t)ETOT * 8 * 4);
  int* srcv   = (int*)alloc((size_t)ETOT * 4);
  int* dstv   = (int*)alloc((size_t)ETOT * 4);
  unsigned* counts = (unsigned*)alloc((size_t)NNODE * 4);
  _Float16* efp  = (_Float16*)alloc((size_t)ETOT * 96 * 2);
  _Float16* hidp = (_Float16*)alloc((size_t)ETOT * MLPD * 2);
  _Float16* w1p  = (_Float16*)alloc((size_t)96 * MLPD * 2);
  _Float16* w2p  = (_Float16*)alloc((size_t)MLPD * 11264 * 2);
  float* stats   = (float*)alloc((size_t)CCH * 4 * 4);
  (void)ws_size; (void)out_size;

  // --- pipeline
  k_pack_a<<<(NNODE * DIN + 255) / 256, 256, 0, stream>>>(x, xp, NNODE, DIN, DIN);
  k_pack_b<<<(DIN * CCH + 255) / 256, 256, 0, stream>>>(embW, ewp, DIN, DIN, CCH);
  k_node_embed<<<256, 128, 0, stream>>>(xp, ewp, embB, hA);
  k_edge_geom<<<(ETOT + 255) / 256, 256, 0, stream>>>(pos, ei, srcv, dstv, shb, radb);
  k_zero_u32<<<(NNODE + 255) / 256, 256, 0, stream>>>(counts, NNODE);
  k_count<<<(ETOT + 255) / 256, 256, 0, stream>>>(dstv, counts);

  float* cur = hA;
  float* nxt = hC;
  for (int li = 0; li < 2; ++li) {
    const int numel  = li ? 11264 : 3072;
    const int npaths = li ? 11 : 3;
    const int nl     = li ? 1 : 3;
    const float i0 = li ? 0.57735027f : 1.0f;   // 1/sqrt(npaths per l3)
    const float i1 = li ? 0.5f : 1.0f;
    const float i2 = li ? 0.5f : 1.0f;

    k_edge_feat<<<(ETOT * 96 + 255) / 256, 256, 0, stream>>>(radb, cur, srcv, dstv, efp);
    k_pack_b<<<(96 * MLPD + 255) / 256, 256, 0, stream>>>(W1[li], w1p, 72, 96, MLPD);
    k_mlp1<<<4096, 128, 0, stream>>>(efp, w1p, B1[li], hidp);
    k_pack_b<<<(MLPD * numel + 255) / 256, 256, 0, stream>>>(W2[li], w2p, MLPD, MLPD, numel);
    k_zero_f32<<<(NNODE * CCH * NCMP + 255) / 256, 256, 0, stream>>>(msgs, NNODE * CCH * NCMP);
    k_mlp2_tpconv<<<ETOT / 16, 128, 0, stream>>>(hidp, w2p, B2p[li], cur, shb, srcv,
                                                 dstv, msgs, numel, npaths);
    k_agg_finish<<<(NNODE * CCH * NCMP + 255) / 256, 256, 0, stream>>>(msgs, counts, i0, i1, i2);
    k_bn_stats<<<CCH, 256, 0, stream>>>(msgs, stats);
    k_bn_apply<<<(NNODE * CCH + 255) / 256, 256, 0, stream>>>(msgs, stats, 3);
    k_product<<<NNODE, 32, 0, stream>>>(msgs, cur, PW1[li], CW2[li], CW3[li], LIN[li],
                                        nxt, nl, li == 0 ? 1 : 0);
    k_bn_stats<<<CCH, 256, 0, stream>>>(nxt, stats);
    k_bn_apply<<<(NNODE * CCH + 255) / 256, 256, 0, stream>>>(nxt, stats, nl);
    float* t = cur; cur = nxt; nxt = t;
  }

  float* out = (float*)d_out;
  k_final_node<<<(NNODE * CCH + 255) / 256, 256, 0, stream>>>(cur, out);
  k_zero_f32<<<(1024 + 255) / 256, 256, 0, stream>>>(out + NNODE * CCH, 1024);
  k_final_graph<<<(NNODE * CCH + 255) / 256, 256, 0, stream>>>(cur, batch, out);
}
